// Encoder_SelfAttention_70360154243484
// MI455X (gfx1250) — matrix operand order repaired and checked by reading, not compiled, not run on hardware
//
#include <hip/hip_runtime.h>
#include <hip/hip_bf16.h>
#include <math.h>

// ---------------------------------------------------------------------------
// Encoder self-attention + analytic order/distance biases + FFT filter layer
// for MI455X (gfx1250, wave32, WMMA).
//
// - All matmul-shaped work -> v_wmma_f32_16x16x32_bf16 (bf16 operands, f32 acc)
// - rfft/filter/irfft == two GEMMs against on-device DFT bases
// - attention keeps scores(f32) + probs(bf16) for a (b,h,64-row) tile in the
//   WGP's 320KB LDS (dynamic LDS ~218KB); K/V/Q tiles staged with CDNA5
//   global_load_async_to_lds_b128 (ASYNCcnt)
// - QKV GEMM writes bf16 directly (consumer precision), halving q/k/v traffic
// ---------------------------------------------------------------------------

typedef __attribute__((ext_vector_type(16))) __bf16 v16bf;
typedef __attribute__((ext_vector_type(8)))  float  v8f;
typedef __attribute__((ext_vector_type(4)))  float  f32x4;
typedef __attribute__((ext_vector_type(4)))  __bf16 bf16x4;

#define WMMA_BF16(a, b, c) \
  __builtin_amdgcn_wmma_f32_16x16x32_bf16(false, (a), false, (b), (short)0, (c), false, false)

// ---- WMMA fragment loaders (contiguous K = 16*g + e order in A and B, wave32) --------------------
// A 16x32 bf16, row-major source: lane = 16*g + m ; element e:
//   K = 16*g + e
__device__ __forceinline__ v16bf load_a_frag(const __bf16* p, int ld) {
  int lane = threadIdx.x & 31;
  int m = lane & 15, g = lane >> 4;
  v16bf r;
#pragma unroll
  for (int e = 0; e < 16; ++e) {
    int j = e >> 1, pb = e & 1;
    int k = g * 16 + e; (void)j; (void)pb;
    r[e] = p[m * ld + k];
  }
  return r;
}

// B 32x16 bf16 from row-major B[k][n]: lane = 16*g + n ; K = 16*g + e
__device__ __forceinline__ v16bf load_b_frag(const __bf16* p, int ld) {
  int lane = threadIdx.x & 31;
  int n = lane & 15, g = lane >> 4;
  v16bf r;
#pragma unroll
  for (int e = 0; e < 16; ++e) r[e] = p[(g * 16 + e) * ld + n];
  return r;
}

// B 32x16 where source is stored transposed (T[n][k], e.g. K-matrix rows):
__device__ __forceinline__ v16bf load_bT_frag(const __bf16* p, int ld) {
  int lane = threadIdx.x & 31;
  int n = lane & 15, g = lane >> 4;
  v16bf r;
#pragma unroll
  for (int e = 0; e < 16; ++e) r[e] = p[n * ld + g * 16 + e];
  return r;
}

__device__ __forceinline__ bf16x4 cvt4(f32x4 v) {
  bf16x4 r;
  r.x = (__bf16)v.x; r.y = (__bf16)v.y; r.z = (__bf16)v.z; r.w = (__bf16)v.w;
  return r;
}

// ---------------------------------------------------------------------------
// Generic GEMM: A[M,K](f32) @ B[K,N](f32) (+bias) -> C(f32) and/or Cb(bf16).
// Block = 128 threads (4 waves), tile 64x64x32. M%64==0, K%32==0; N guarded
// at float4 granularity (N%4==0 in all uses).
// ---------------------------------------------------------------------------
__global__ void gemm_wmma(const float* __restrict__ A, const float* __restrict__ B,
                          const float* __restrict__ bias, float* __restrict__ C,
                          __bf16* __restrict__ Cb, int M, int N, int K) {
  __shared__ __bf16 As[64][40];   // 80B rows: 16 consecutive rows hit distinct banks
  __shared__ __bf16 Bs[32][72];
  int tid = threadIdx.x;
  int wave = tid >> 5;
  int m0 = blockIdx.y * 64;
  int n0 = blockIdx.x * 64;
  int wm = (wave & 1) * 32;
  int wn = (wave >> 1) * 32;
  v8f acc[2][2] = {};

  for (int k0 = 0; k0 < K; k0 += 32) {
    if (k0 + 32 < K) __builtin_prefetch(&A[(size_t)(m0 + (tid >> 1)) * K + k0 + 32], 0, 1);
    // A tile 64x32: 512 float4 chunks, 4 per thread -> global_load_b128 + ds_store_b64
#pragma unroll
    for (int i = 0; i < 4; ++i) {
      int q4 = tid + i * 128;               // 0..511
      int r = q4 >> 3, c = (q4 & 7) * 4;
      f32x4 av = *(const f32x4*)&A[(size_t)(m0 + r) * K + k0 + c];
      *(bf16x4*)&As[r][c] = cvt4(av);
    }
    // B tile 32x64: 512 float4 chunks, guarded as whole vectors
#pragma unroll
    for (int i = 0; i < 4; ++i) {
      int q4 = tid + i * 128;
      int r = q4 >> 4, c = (q4 & 15) * 4;
      int col = n0 + c;
      f32x4 bv = {0.0f, 0.0f, 0.0f, 0.0f};
      if (col < N) bv = *(const f32x4*)&B[(size_t)(k0 + r) * N + col];
      *(bf16x4*)&Bs[r][c] = cvt4(bv);
    }
    __syncthreads();
#pragma unroll
    for (int i = 0; i < 2; ++i) {
      v16bf a = load_a_frag(&As[wm + i * 16][0], 40);
#pragma unroll
      for (int j = 0; j < 2; ++j) {
        v16bf b = load_b_frag(&Bs[0][wn + j * 16], 72);
        acc[i][j] = WMMA_BF16(a, b, acc[i][j]);
      }
    }
    __syncthreads();
  }

  int lane = tid & 31;
  int n = lane & 15, g = lane >> 4;
#pragma unroll
  for (int i = 0; i < 2; ++i)
#pragma unroll
    for (int j = 0; j < 2; ++j)
#pragma unroll
      for (int e = 0; e < 8; ++e) {
        int row = m0 + wm + i * 16 + e + 8 * g;
        int col = n0 + wn + j * 16 + n;
        if (col < N) {
          float v = acc[i][j][e];
          if (bias) v += bias[col];
          if (C)  C[(size_t)row * N + col] = v;
          if (Cb) Cb[(size_t)row * N + col] = (__bf16)v;
        }
      }
}

// ---------------------------------------------------------------------------
// Per-(b,h,s) scalar projections for the order/distance biases (bf16 q/k in).
// ---------------------------------------------------------------------------
__global__ void proj_kernel(const __bf16* __restrict__ q, const __bf16* __restrict__ k,
                            const float* __restrict__ Wo, const float* __restrict__ Wd,
                            float* __restrict__ oq, float* __restrict__ ok,
                            float* __restrict__ dq, float* __restrict__ dk) {
  int id = blockIdx.x * blockDim.x + threadIdx.x;
  if (id >= 8 * 8 * 512) return;
  int s = id & 511, h = (id >> 9) & 7, b = id >> 12;
  const __bf16* qr = q + (size_t)(b * 512 + s) * 512 + h * 64;
  const __bf16* kr = k + (size_t)(b * 512 + s) * 512 + h * 64;
  float a0 = 0, a1 = 0, a2 = 0, a3 = 0;
#pragma unroll 8
  for (int d = 0; d < 64; ++d) {
    float qv = (float)qr[d], kv = (float)kr[d];
    a0 += qv * Wo[d];      a1 += kv * Wo[64 + d];
    a2 += qv * Wd[d];      a3 += kv * Wd[64 + d];
  }
  int idx = (b * 8 + h) * 512 + s;
  oq[idx] = a0; ok[idx] = a1; dq[idx] = a2; dk[idx] = a3;
}

// ---------------------------------------------------------------------------
// CDNA5 async global->LDS staging of a 64x64 bf16 tile (row stride 512 elems)
// into LDS with ld=72 (144B rows, 16B aligned). 256 threads, 2 x 16B each.
// Tracked with ASYNCcnt; waited before the workgroup barrier.
// ---------------------------------------------------------------------------
__device__ __forceinline__ void stage_tile_async(__bf16* dst, const __bf16* src, int tid) {
#pragma unroll
  for (int i = 0; i < 2; ++i) {
    int chunk = tid + i * 256;            // 0..511
    int r = chunk >> 3, c = (chunk & 7) * 8;
    const __bf16* g = src + (size_t)r * 512 + c;
    unsigned lds = (unsigned)(size_t)(dst + r * 72 + c);
    asm volatile("global_load_async_to_lds_b128 %0, %1, off"
                 :: "v"(lds), "v"(g) : "memory");
  }
}
__device__ __forceinline__ void wait_async_zero() {
  asm volatile("s_wait_asynccnt 0" ::: "memory");
}

// ---------------------------------------------------------------------------
// Attention: one workgroup per (b, h, 64-query-row tile). 256 threads, 8 waves.
// Dynamic LDS (~218 KB of the WGP's 320 KB).
// ---------------------------------------------------------------------------
#define SC_LD 516
#define PR_LD 520
#define ATTN_SC_BYTES (64 * SC_LD * 4)
#define ATTN_PR_BYTES (64 * PR_LD * 2)
#define ATTN_QS_BYTES (64 * 72 * 2)
#define ATTN_KV_BYTES (64 * 72 * 2)
#define ATTN_F32_CNT  (512 + 512 + 64 + 64 + 256 + 64 + 64)
#define ATTN_LDS_BYTES (ATTN_SC_BYTES + ATTN_PR_BYTES + ATTN_QS_BYTES + ATTN_KV_BYTES + ATTN_F32_CNT * 4)

__global__ void attn_kernel(const __bf16* __restrict__ q, const __bf16* __restrict__ k,
                            const __bf16* __restrict__ v,
                            const float* __restrict__ oq, const float* __restrict__ ok,
                            const float* __restrict__ dq, const float* __restrict__ dk,
                            const float* __restrict__ b_order, const float* __restrict__ b_dist,
                            const float* __restrict__ scalar, const float* __restrict__ mask,
                            float* __restrict__ ctx) {
  extern __shared__ char smem[];
  float*  sc    = (float*)smem;
  __bf16* probs = (__bf16*)(smem + ATTN_SC_BYTES);
  __bf16* qs    = (__bf16*)(smem + ATTN_SC_BYTES + ATTN_PR_BYTES);
  __bf16* kvt   = qs + 64 * 72;
  float*  fbp   = (float*)(smem + ATTN_SC_BYTES + ATTN_PR_BYTES + ATTN_QS_BYTES + ATTN_KV_BYTES);
  float* okv = fbp;          float* dkv = okv + 512;
  float* oqv = dkv + 512;    float* dqv = oqv + 64;
  float* red = dqv + 64;     float* rowm = red + 256;   float* rows = rowm + 64;

  const int S = 512, HID = 512, D = 64;
  int tid = threadIdx.x;
  int b = blockIdx.z, h = blockIdx.y, q0 = blockIdx.x * 64;
  int bh = b * 8 + h;

  // stage q tile (async) + bias projection vectors
  stage_tile_async(qs, q + (size_t)(b * S + q0) * HID + h * D, tid);
  for (int i = tid; i < S; i += 256) { okv[i] = ok[bh * S + i]; dkv[i] = dk[bh * S + i]; }
  for (int i = tid; i < 64; i += 256) { oqv[i] = oq[bh * S + q0 + i]; dqv[i] = dq[bh * S + q0 + i]; }
  wait_async_zero();
  __syncthreads();

  int wave = tid >> 5, lane = tid & 31;
  int ln = lane & 15, g = lane >> 4;

  // ---- Phase 1: raw scores = q @ k^T into LDS ----
  for (int kb = 0; kb < 8; ++kb) {
    stage_tile_async(kvt, k + (size_t)(b * S + kb * 64) * HID + h * D, tid);
    wait_async_zero();
    __syncthreads();
#pragma unroll
    for (int t = 0; t < 2; ++t) {
      int tile = wave * 2 + t;
      int qi = tile >> 2, ni = tile & 3;
      v8f c = {};
#pragma unroll
      for (int kk = 0; kk < 2; ++kk) {
        v16bf a = load_a_frag(&qs[qi * 16 * 72 + kk * 32], 72);
        v16bf bfr = load_bT_frag(&kvt[ni * 16 * 72 + kk * 32], 72);
        c = WMMA_BF16(a, bfr, c);
      }
#pragma unroll
      for (int e = 0; e < 8; ++e)
        sc[(qi * 16 + e + 8 * g) * SC_LD + kb * 64 + ni * 16 + ln] = c[e];
    }
    __syncthreads();
  }

  // ---- Phase 2: biases + softmax (4 threads per row) ----
  int row = tid >> 2, quad = tid & 3;
  float bo = b_order[0], bd = b_dist[0];
  float sc2 = scalar[0] * scalar[0] * 0.5f;
  int ig = q0 + row;
  float oqr = oqv[row], dqr = dqv[row];
  float mx = -3.0e38f;
  for (int c = quad * 128; c < quad * 128 + 128; ++c) {
    float raw = sc[row * SC_LD + c];
    float oo = oqr + okv[c] + bo;
    float pr = 1.0f / (1.0f + expf(-oo));
    float eo = (c > ig) ? logf(pr + 1e-24f) : logf(1.0f - pr + 1e-24f);
    float dd = dqr + dkv[c] + bd;
    float gd = logf(fabsf((float)(c - ig)) + 1.0f);
    float ed = -(gd - dd) * (gd - dd) * sc2;
    float sv = (raw + eo + ed) * 0.125f + mask[b * S + c];   // 0.125 = 1/sqrt(64)
    sc[row * SC_LD + c] = sv;
    mx = fmaxf(mx, sv);
  }
  red[tid] = mx;
  __syncthreads();
  if (quad == 0)
    rowm[row] = fmaxf(fmaxf(red[tid], red[tid + 1]), fmaxf(red[tid + 2], red[tid + 3]));
  __syncthreads();
  float rm = rowm[row], sum = 0.0f;
  for (int c = quad * 128; c < quad * 128 + 128; ++c) {
    float e = expf(sc[row * SC_LD + c] - rm);
    sc[row * SC_LD + c] = e;
    sum += e;
  }
  red[tid] = sum;
  __syncthreads();
  if (quad == 0)
    rows[row] = red[tid] + red[tid + 1] + red[tid + 2] + red[tid + 3];
  __syncthreads();
  float inv = 1.0f / rows[row];
  for (int c = quad * 128; c < quad * 128 + 128; ++c)
    probs[row * PR_LD + c] = (__bf16)(sc[row * SC_LD + c] * inv);
  __syncthreads();

  // ---- Phase 3: ctx = probs @ v ----
  v8f acc[2] = {};
  for (int vb = 0; vb < 8; ++vb) {
    stage_tile_async(kvt, v + (size_t)(b * S + vb * 64) * HID + h * D, tid);
    wait_async_zero();
    __syncthreads();
#pragma unroll
    for (int t = 0; t < 2; ++t) {
      int tile = wave * 2 + t;
      int qi = tile >> 2, ni = tile & 3;
#pragma unroll
      for (int kk = 0; kk < 2; ++kk) {
        v16bf a = load_a_frag(&probs[qi * 16 * PR_LD + vb * 64 + kk * 32], PR_LD);
        v16bf bfr = load_b_frag(&kvt[kk * 32 * 72 + ni * 16], 72);
        acc[t] = WMMA_BF16(a, bfr, acc[t]);
      }
    }
    __syncthreads();
  }
#pragma unroll
  for (int t = 0; t < 2; ++t) {
    int tile = wave * 2 + t;
    int qi = tile >> 2, ni = tile & 3;
#pragma unroll
    for (int e = 0; e < 8; ++e)
      ctx[(size_t)(b * S + q0 + qi * 16 + e + 8 * g) * HID + h * D + ni * 16 + ln] = acc[t][e];
  }
}

// ---------------------------------------------------------------------------
// DFT bases (ortho). Forward: 512 x 544 (cols 0..271 Re, 272..543 Im, f<=256).
// Inverse: 544 x 512 with the 2x / DC / Nyquist coefficients folded in.
// Integer-mod angle reduction keeps sin/cos args small and exact.
// ---------------------------------------------------------------------------
#define INV_SQRT512 0.044194173824159216f
#define TWO_PI_OVER_512 0.012271846303085130f

__global__ void build_fwd_basis(float* __restrict__ Bf) {
  int id = blockIdx.x * blockDim.x + threadIdx.x;
  if (id >= 512 * 544) return;
  int n = id / 544, c = id % 544;
  int f = (c < 272) ? c : c - 272;
  float val = 0.0f;
  if (f <= 256) {
    int fn = (f * n) & 511;
    float th = TWO_PI_OVER_512 * (float)fn;
    val = (c < 272) ? cosf(th) * INV_SQRT512 : -sinf(th) * INV_SQRT512;
  }
  Bf[id] = val;
}

__global__ void build_inv_basis(float* __restrict__ Bi) {
  int id = blockIdx.x * blockDim.x + threadIdx.x;
  if (id >= 544 * 512) return;
  int r = id / 512, n = id % 512;
  int f = (r < 272) ? r : r - 272;
  float val = 0.0f;
  if (f <= 256) {
    float a = (f == 0 || f == 256) ? 1.0f : 2.0f;
    int fn = (f * n) & 511;
    float th = TWO_PI_OVER_512 * (float)fn;
    val = (r < 272) ? a * cosf(th) * INV_SQRT512 : -a * sinf(th) * INV_SQRT512;
  }
  Bi[id] = val;
}

// Elementwise complex multiply by the learned filter: Y = X * w  (per (s,f)).
__global__ void filter_kernel(const float* __restrict__ spec, const float* __restrict__ cw,
                              float* __restrict__ ysp) {
  int id = blockIdx.x * blockDim.x + threadIdx.x;
  if (id >= 4096 * 272) return;
  int m = id / 272, f = id % 272;
  float yr = 0.0f, yi = 0.0f;
  if (f <= 256) {
    int s = m & 511;
    float R = spec[(size_t)m * 544 + f];
    float I = spec[(size_t)m * 544 + 272 + f];
    float wr = cw[(s * 257 + f) * 2];
    float wi = cw[(s * 257 + f) * 2 + 1];
    yr = R * wr - I * wi;
    yi = R * wi + I * wr;
  }
  ysp[(size_t)m * 544 + f] = yr;
  ysp[(size_t)m * 544 + 272 + f] = yi;
}

// ---------------------------------------------------------------------------
// Double layernorm epilogue: out = LN(LN(seq_fft + ctx) + input). 1 block/row.
// ---------------------------------------------------------------------------
__global__ void epilogue_kernel(const float* __restrict__ seq, const float* __restrict__ ctx,
                                const float* __restrict__ inp,
                                const float* __restrict__ lfw, const float* __restrict__ lfb,
                                const float* __restrict__ lw, const float* __restrict__ lb,
                                float* __restrict__ out) {
  __shared__ float red[128];
  __shared__ float stats[2];
  int m = blockIdx.x, tid = threadIdx.x;
  float t[4];
  float sum = 0.0f;
#pragma unroll
  for (int i = 0; i < 4; ++i) {
    int c = tid + i * 128;
    t[i] = seq[(size_t)m * 512 + c] + ctx[(size_t)m * 512 + c];
    sum += t[i];
  }
  red[tid] = sum; __syncthreads();
  for (int s = 64; s > 0; s >>= 1) { if (tid < s) red[tid] += red[tid + s]; __syncthreads(); }
  if (tid == 0) stats[0] = red[0] * (1.0f / 512.0f);
  __syncthreads();
  float mean = stats[0];
  float vs = 0.0f;
#pragma unroll
  for (int i = 0; i < 4; ++i) { float d = t[i] - mean; vs += d * d; }
  red[tid] = vs; __syncthreads();
  for (int s = 64; s > 0; s >>= 1) { if (tid < s) red[tid] += red[tid + s]; __syncthreads(); }
  if (tid == 0) stats[1] = red[0] * (1.0f / 512.0f);
  __syncthreads();
  float rstd = rsqrtf(stats[1] + 1e-12f);

  float t2[4];
  sum = 0.0f;
#pragma unroll
  for (int i = 0; i < 4; ++i) {
    int c = tid + i * 128;
    float hg = lfw[c] * ((t[i] - mean) * rstd) + lfb[c];
    t2[i] = hg + inp[(size_t)m * 512 + c];
    sum += t2[i];
  }
  red[tid] = sum; __syncthreads();
  for (int s = 64; s > 0; s >>= 1) { if (tid < s) red[tid] += red[tid + s]; __syncthreads(); }
  if (tid == 0) stats[0] = red[0] * (1.0f / 512.0f);
  __syncthreads();
  float mean2 = stats[0];
  vs = 0.0f;
#pragma unroll
  for (int i = 0; i < 4; ++i) { float d = t2[i] - mean2; vs += d * d; }
  red[tid] = vs; __syncthreads();
  for (int s = 64; s > 0; s >>= 1) { if (tid < s) red[tid] += red[tid + s]; __syncthreads(); }
  if (tid == 0) stats[1] = red[0] * (1.0f / 512.0f);
  __syncthreads();
  float rstd2 = rsqrtf(stats[1] + 1e-12f);
#pragma unroll
  for (int i = 0; i < 4; ++i) {
    int c = tid + i * 128;
    out[(size_t)m * 512 + c] = lw[c] * ((t2[i] - mean2) * rstd2) + lb[c];
  }
}

// ---------------------------------------------------------------------------
extern "C" void kernel_launch(void* const* d_in, const int* in_sizes, int n_in,
                              void* d_out, int out_size, void* d_ws, size_t ws_size,
                              hipStream_t stream) {
  (void)in_sizes; (void)n_in; (void)out_size; (void)ws_size;
  const float* x    = (const float*)d_in[0];
  const float* mask = (const float*)d_in[1];
  const float* Wq   = (const float*)d_in[2];
  const float* bq   = (const float*)d_in[3];
  const float* Wk   = (const float*)d_in[4];
  const float* bk   = (const float*)d_in[5];
  const float* Wv   = (const float*)d_in[6];
  const float* bv   = (const float*)d_in[7];
  const float* Wo   = (const float*)d_in[8];
  const float* b_o  = (const float*)d_in[9];
  const float* Wd   = (const float*)d_in[10];
  const float* b_d  = (const float*)d_in[11];
  const float* scl  = (const float*)d_in[12];
  const float* cw   = (const float*)d_in[13];
  const float* lfw  = (const float*)d_in[14];
  const float* lfb  = (const float*)d_in[15];
  const float* lw   = (const float*)d_in[16];
  const float* lb   = (const float*)d_in[17];
  float* out = (float*)d_out;

  char* ws = (char*)d_ws;
  size_t off = 0;
  auto carve = [&](size_t bytes) -> char* {
    char* p = ws + off;
    off += (bytes + 255) & ~(size_t)255;
    return p;
  };
  __bf16* qb  = (__bf16*)carve(4096ull * 512 * 2);
  __bf16* kb  = (__bf16*)carve(4096ull * 512 * 2);
  __bf16* vb  = (__bf16*)carve(4096ull * 512 * 2);
  float* ctx  = (float*)carve(4096ull * 512 * 4);
  float* oq   = (float*)carve(32768ull * 4);
  float* okp  = (float*)carve(32768ull * 4);
  float* dq   = (float*)carve(32768ull * 4);
  float* dk   = (float*)carve(32768ull * 4);
  float* Bf   = (float*)carve(512ull * 544 * 4);
  float* Bi   = (float*)carve(544ull * 512 * 4);
  float* spec = (float*)carve(4096ull * 544 * 4);
  float* ysp  = (float*)carve(4096ull * 544 * 4);
  float* sfft = (float*)carve(4096ull * 512 * 4);

  // Allow the big dynamic-LDS attention kernel (>64KB, WGP has 320KB).
  (void)hipFuncSetAttribute(reinterpret_cast<const void*>(attn_kernel),
                            hipFuncAttributeMaxDynamicSharedMemorySize, ATTN_LDS_BYTES);

  dim3 gqkv(8, 64);
  gemm_wmma<<<gqkv, 128, 0, stream>>>(x, Wq, bq, nullptr, qb, 4096, 512, 512);
  gemm_wmma<<<gqkv, 128, 0, stream>>>(x, Wk, bk, nullptr, kb, 4096, 512, 512);
  gemm_wmma<<<gqkv, 128, 0, stream>>>(x, Wv, bv, nullptr, vb, 4096, 512, 512);

  proj_kernel<<<128, 256, 0, stream>>>(qb, kb, Wo, Wd, oq, okp, dq, dk);
  build_fwd_basis<<<(512 * 544 + 255) / 256, 256, 0, stream>>>(Bf);
  build_inv_basis<<<(544 * 512 + 255) / 256, 256, 0, stream>>>(Bi);

  attn_kernel<<<dim3(8, 8, 8), 256, ATTN_LDS_BYTES, stream>>>(
      qb, kb, vb, oq, okp, dq, dk, b_o, b_d, scl, mask, ctx);

  gemm_wmma<<<dim3(9, 64), 128, 0, stream>>>(ctx, Bf, nullptr, spec, nullptr, 4096, 544, 512);
  filter_kernel<<<(4096 * 272 + 255) / 256, 256, 0, stream>>>(spec, cw, ysp);
  gemm_wmma<<<dim3(8, 64), 128, 0, stream>>>(ysp, Bi, nullptr, sfft, nullptr, 4096, 512, 544);

  epilogue_kernel<<<4096, 128, 0, stream>>>(sfft, ctx, x, lfw, lfb, lw, lb, out);
}
